// MyModel_87522843560443
// MI455X (gfx1250) — matrix operand order repaired and checked by reading, not compiled, not run on hardware
//
#include <hip/hip_runtime.h>

// ---------------------------------------------------------------------------
// Model dims
// ---------------------------------------------------------------------------
#define Bb   128
#define Tt   300
#define TCc  20
#define DWw  16
#define DCc  32
#define UCc  32
#define UMm  64
#define POSp 20
#define PARp 8
#define VWw  1834
#define VCc  132
#define DINd 76          // DW + UC + POS + PAR
#define XK   96          // DIN padded to 32-multiple
#define WKF  160         // word LSTM K: 96 (x, padded) + 64 (h)
#define NSEQ (Bb * Tt)   // 38400

typedef __attribute__((ext_vector_type(16))) _Float16 v16h;
typedef __attribute__((ext_vector_type(8)))  float    v8f;
typedef __attribute__((ext_vector_type(4)))  unsigned int u32x4;

union FragU { v16h h; u32x4 u[2]; };

// ---------------------------------------------------------------------------
// WMMA fragment loaders (CDNA5 16-bit A/B lane layouts, ISA 7.12.2).
// A 16x32 f16: lane<16 holds K dword-pairs {0..3, 8..11}; lane>=16 holds
// {4..7, 12..15}. With row-major K-contiguous storage that is two 16B chunks.
// B 32x16 f16: same K order as A (lane<16: K {0..7,16..23}; lane>=16:
// {8..15,24..31}), column N = lane&15 -> store W transposed [N][K].
// ---------------------------------------------------------------------------
__device__ __attribute__((always_inline)) inline
v16h load_fragA(const _Float16* rowKT, int laneHi) {
    FragU f;
    const _Float16* p = rowKT + (laneHi ? 8 : 0);
    f.u[0] = *(const u32x4*)(p);
    f.u[1] = *(const u32x4*)(p + 16);
    return f.h;
}

__device__ __attribute__((always_inline)) inline
v16h load_fragB(const _Float16* colKT, int laneHi) {
    FragU f;
    const _Float16* p = colKT + (laneHi ? 8 : 0);
    f.u[0] = *(const u32x4*)(p);
    f.u[1] = *(const u32x4*)(p + 16);
    return f.h;
}

__device__ __attribute__((always_inline)) inline
v8f wmma_f16(v16h a, v16h b, v8f c) {
    return __builtin_amdgcn_wmma_f32_16x16x32_f16(false, a, false, b,
                                                  (short)0, c, false, false);
}

// Fast transcendentals: v_exp_f32 + v_rcp_f32, no IEEE-divide expansion.
__device__ __attribute__((always_inline)) inline float fast_rcp(float x) {
    return __builtin_amdgcn_rcpf(x);
}
__device__ __attribute__((always_inline)) inline float sigm(float x) {
    return fast_rcp(1.0f + __expf(-x));
}
__device__ __attribute__((always_inline)) inline float tanh_f(float x) {
    // tanh(x) = 2 / (1 + e^{-2x}) - 1
    return __builtin_fmaf(2.0f, fast_rcp(1.0f + __expf(-2.0f * x)), -1.0f);
}

// ---------------------------------------------------------------------------
// prep: f32 -> f16 conversions; weights transposed to [N][Kpad] layout.
// charWT[n][k]: k<32 -> char_Wx[k][n], else char_Wh[k-32][n]        (128 x 64)
// fwd/bwdWT[n][k]: k<76 Wx[k][n]; 76..95 zero pad; 96..159 Wh[k-96][n] (256x160)
// ---------------------------------------------------------------------------
__global__ void prep_kernel(const float* __restrict__ emb_char,
                            const float* __restrict__ cWx, const float* __restrict__ cWh,
                            const float* __restrict__ fWx, const float* __restrict__ fWh,
                            const float* __restrict__ bWx, const float* __restrict__ bWh,
                            _Float16* __restrict__ embH, _Float16* __restrict__ charWT,
                            _Float16* __restrict__ fwdWT, _Float16* __restrict__ bwdWT) {
    int tid = blockIdx.x * blockDim.x + threadIdx.x;
    int stride = gridDim.x * blockDim.x;
    for (int i = tid; i < VCc * DCc; i += stride)
        embH[i] = (_Float16)emb_char[i];
    for (int i = tid; i < 128 * 64; i += stride) {
        int n = i >> 6, k = i & 63;
        float v = (k < 32) ? cWx[k * 128 + n] : cWh[(k - 32) * 128 + n];
        charWT[i] = (_Float16)v;
    }
    for (int i = tid; i < 256 * WKF; i += stride) {
        int n = i / WKF, k = i % WKF;
        float vf, vb;
        if (k < DINd)      { vf = fWx[k * 256 + n];        vb = bWx[k * 256 + n]; }
        else if (k < XK)   { vf = 0.0f;                    vb = 0.0f; }
        else               { vf = fWh[(k - XK) * 256 + n]; vb = bWh[(k - XK) * 256 + n]; }
        fwdWT[i] = (_Float16)vf;
        bwdWT[i] = (_Float16)vb;
    }
}

// ---------------------------------------------------------------------------
// build_X: X[s][0:16]=emb_wor[word], [48:68]=pos, [68:76]=par, [76:96]=0.
// [16:48] (char-LSTM h) is written by char_lstm_kernel.
// ---------------------------------------------------------------------------
__global__ void build_x_kernel(const int* __restrict__ word_in,
                               const float* __restrict__ pos,
                               const float* __restrict__ par,
                               const float* __restrict__ emb_wor,
                               _Float16* __restrict__ X) {
    int s = blockIdx.x * blockDim.x + threadIdx.x;
    if (s >= NSEQ) return;
    _Float16* xp = X + (size_t)s * XK;
    const float* e = emb_wor + (size_t)word_in[s] * DWw;
    #pragma unroll
    for (int i = 0; i < DWw; ++i) xp[i] = (_Float16)e[i];
    const float* pp = pos + (size_t)s * POSp;
    #pragma unroll
    for (int i = 0; i < POSp; ++i) xp[48 + i] = (_Float16)pp[i];
    const float* qq = par + (size_t)s * PARp;
    #pragma unroll
    for (int i = 0; i < PARp; ++i) xp[68 + i] = (_Float16)qq[i];
    #pragma unroll
    for (int i = DINd; i < XK; ++i) xp[i] = (_Float16)0.0f;
}

// ---------------------------------------------------------------------------
// Char LSTM: 1 wave = 16 sequences; 8 waves/block; 300 blocks.
// z(16x128) = x_t(16x32)@Wx + h(16x32)@Wh + b via 16 WMMAs/step.
// B frags persistent in VGPRs; h recirculated through per-wave LDS tile.
// Final h (carried, Keras mask semantics) written into X[:,16:48] as f16.
// ---------------------------------------------------------------------------
__global__ __launch_bounds__(256) void char_lstm_kernel(
        const int* __restrict__ char_in, const float* __restrict__ char_b,
        const _Float16* __restrict__ embH, const _Float16* __restrict__ charWT,
        _Float16* __restrict__ X) {
    __shared__ __align__(32) _Float16 hbuf[8][16][32];

    const int lane   = threadIdx.x & 31;
    const int wave   = threadIdx.x >> 5;
    const int laneHi = lane >> 4;
    const int l15    = lane & 15;
    const int seqBase = blockIdx.x * 128 + wave * 16;

    for (int i = threadIdx.x; i < 8 * 16 * 32; i += 256)
        ((_Float16*)hbuf)[i] = (_Float16)0.0f;
    __syncthreads();

    // Persistent B fragments: 8 N-tiles x {Wx, Wh}; bias splat as v8f frags.
    v16h Bx[8], Bh[8];
    v8f  biasF[8];
    #pragma unroll
    for (int n = 0; n < 8; ++n) {
        const _Float16* colBase = charWT + (size_t)(n * 16 + l15) * 64;
        Bx[n] = load_fragB(colBase,      laneHi);
        Bh[n] = load_fragB(colBase + 32, laneHi);
        const float bv = char_b[n * 16 + l15];
        #pragma unroll
        for (int r = 0; r < 8; ++r) biasF[n][r] = bv;
    }

    float cs[2][8], hs[2][8];
    #pragma unroll
    for (int ut = 0; ut < 2; ++ut)
        #pragma unroll
        for (int r = 0; r < 8; ++r) { cs[ut][r] = 0.0f; hs[ut][r] = 0.0f; }

    const int myRow = seqBase + l15;

    for (int t = 0; t < TCc; ++t) {
        const int id = char_in[(size_t)myRow * TCc + t];

        v16h A0 = load_fragA(embH + (size_t)id * DCc, laneHi);     // x_t
        v16h A1 = load_fragA(&hbuf[wave][l15][0], laneHi);         // h_{t-1}

        v8f acc[8];
        #pragma unroll
        for (int n = 0; n < 8; ++n) {
            v8f a = biasF[n];
            a = wmma_f16(A0, Bx[n], a);
            a = wmma_f16(A1, Bh[n], a);
            acc[n] = a;
        }

        // gates: i = tiles 0..1, f = 2..3, g = 4..5, o = 6..7
        #pragma unroll
        for (int ut = 0; ut < 2; ++ut) {
            #pragma unroll
            for (int r = 0; r < 8; ++r) {
                const int Mrow = r + laneHi * 8;
                const int idr  = __shfl(id, Mrow, 32);
                float iv = sigm(acc[ut][r]);
                float fv = sigm(acc[2 + ut][r]);
                float gv = tanh_f(acc[4 + ut][r]);
                float ov = sigm(acc[6 + ut][r]);
                float cn = fv * cs[ut][r] + iv * gv;
                float hn = ov * tanh_f(cn);
                bool  m  = (idr != 0);
                cs[ut][r] = m ? cn : cs[ut][r];
                hs[ut][r] = m ? hn : hs[ut][r];
                hbuf[wave][Mrow][ut * 16 + l15] = (_Float16)hs[ut][r];
            }
        }
    }

    // write carried final h into X[:, 16:48]
    #pragma unroll
    for (int ut = 0; ut < 2; ++ut)
        #pragma unroll
        for (int r = 0; r < 8; ++r) {
            const int Mrow = r + laneHi * 8;
            const size_t s = (size_t)(seqBase + Mrow);
            X[s * XK + DWw + ut * 16 + l15] = (_Float16)hs[ut][r];
        }
}

// ---------------------------------------------------------------------------
// Word BiLSTM: block 0 = fwd, block 1 = bwd; 16 waves/block.
// Wave w: mtile = w>>1 (16 batch rows), gate-aligned ntiles {t,4+t,8+t,12+t}
// for t in {2*(w&1), 2*(w&1)+1} so all 4 gates of its 32 units stay in-lane.
// z(128x256) = [x_t | h](128x160) @ WT per step: 40 WMMAs/wave/step.
// h exchanged via 16KB LDS buffer; 2 barriers/step.
// ---------------------------------------------------------------------------
__global__ __launch_bounds__(512) void word_lstm_kernel(
        const int* __restrict__ word_in,
        const float* __restrict__ fwd_b, const float* __restrict__ bwd_b,
        const _Float16* __restrict__ fwdWT, const _Float16* __restrict__ bwdWT,
        const _Float16* __restrict__ X,
        float* __restrict__ hf, float* __restrict__ hb) {
    __shared__ __align__(32) _Float16 hbuf[128][64];

    const int dir = blockIdx.x;
    const float*    bvec = dir ? bwd_b : fwd_b;
    const _Float16* WT   = dir ? bwdWT : fwdWT;
    float*          hout = dir ? hb    : hf;

    const int lane   = threadIdx.x & 31;
    const int wave   = threadIdx.x >> 5;
    const int laneHi = lane >> 4;
    const int l15    = lane & 15;
    const int mtile  = wave >> 1;
    const int half   = wave & 1;

    for (int i = threadIdx.x; i < 128 * 64; i += 512)
        ((_Float16*)hbuf)[i] = (_Float16)0.0f;

    int ntl[2][4];
    v8f biasF[2][4];
    #pragma unroll
    for (int ut = 0; ut < 2; ++ut) {
        const int tloc = 2 * half + ut;
        #pragma unroll
        for (int g = 0; g < 4; ++g) {
            ntl[ut][g] = g * 4 + tloc;
            const float bv = bvec[ntl[ut][g] * 16 + l15];
            #pragma unroll
            for (int r = 0; r < 8; ++r) biasF[ut][g][r] = bv;
        }
    }

    float cs[2][8], hs[2][8];
    #pragma unroll
    for (int ut = 0; ut < 2; ++ut)
        #pragma unroll
        for (int r = 0; r < 8; ++r) { cs[ut][r] = 0.0f; hs[ut][r] = 0.0f; }

    const int browLane = mtile * 16 + l15;
    __syncthreads();

    for (int s = 0; s < Tt; ++s) {
        const int tt = dir ? (Tt - 1 - s) : s;

        v16h A[5];
        const _Float16* xrow = X + ((size_t)browLane * Tt + tt) * XK;
        A[0] = load_fragA(xrow,       laneHi);
        A[1] = load_fragA(xrow + 32,  laneHi);
        A[2] = load_fragA(xrow + 64,  laneHi);
        A[3] = load_fragA(&hbuf[browLane][0],  laneHi);
        A[4] = load_fragA(&hbuf[browLane][32], laneHi);
        __syncthreads();                 // all h reads done before rewrites

        v8f acc[2][4];
        #pragma unroll
        for (int ut = 0; ut < 2; ++ut)
            #pragma unroll
            for (int g = 0; g < 4; ++g)
                acc[ut][g] = biasF[ut][g];

        #pragma unroll
        for (int kt = 0; kt < 5; ++kt)
            #pragma unroll
            for (int ut = 0; ut < 2; ++ut)
                #pragma unroll
                for (int g = 0; g < 4; ++g) {
                    const int n = ntl[ut][g];
                    v16h Bf = load_fragB(
                        WT + (size_t)(n * 16 + l15) * WKF + kt * 32, laneHi);
                    acc[ut][g] = wmma_f16(A[kt], Bf, acc[ut][g]);
                }

        int wid[8];
        #pragma unroll
        for (int r = 0; r < 8; ++r)
            wid[r] = word_in[(size_t)(mtile * 16 + r + laneHi * 8) * Tt + tt];

        #pragma unroll
        for (int ut = 0; ut < 2; ++ut) {
            const int u = (2 * half + ut) * 16 + l15;
            #pragma unroll
            for (int r = 0; r < 8; ++r) {
                const int Mrow = r + laneHi * 8;
                float iv = sigm(acc[ut][0][r]);
                float fv = sigm(acc[ut][1][r]);
                float gv = tanh_f(acc[ut][2][r]);
                float ov = sigm(acc[ut][3][r]);
                float cn = fv * cs[ut][r] + iv * gv;
                float hn = ov * tanh_f(cn);
                bool  m  = (wid[r] != 0);
                cs[ut][r] = m ? cn : cs[ut][r];
                hs[ut][r] = m ? hn : hs[ut][r];
                const int brow = mtile * 16 + Mrow;
                hbuf[brow][u] = (_Float16)hs[ut][r];
                hout[((size_t)brow * Tt + tt) * UMm + u] = hs[ut][r];
            }
        }
        __syncthreads();                 // writes visible before next reads
    }
}

// ---------------------------------------------------------------------------
// Dense(128->4) + softmax, one thread per (b,t).
// ---------------------------------------------------------------------------
__global__ void dense_softmax_kernel(const float* __restrict__ hf,
                                     const float* __restrict__ hb,
                                     const float* __restrict__ W,
                                     const float* __restrict__ b,
                                     float* __restrict__ out) {
    int idx = blockIdx.x * blockDim.x + threadIdx.x;
    if (idx >= NSEQ) return;
    float z[4];
    #pragma unroll
    for (int k = 0; k < 4; ++k) z[k] = b[k];
    const float* hfp = hf + (size_t)idx * UMm;
    const float* hbp = hb + (size_t)idx * UMm;
    for (int j = 0; j < UMm; ++j) {
        float v = hfp[j];
        #pragma unroll
        for (int k = 0; k < 4; ++k) z[k] += v * W[j * 4 + k];
    }
    for (int j = 0; j < UMm; ++j) {
        float v = hbp[j];
        #pragma unroll
        for (int k = 0; k < 4; ++k) z[k] += v * W[(UMm + j) * 4 + k];
    }
    float m = z[0];
    #pragma unroll
    for (int k = 1; k < 4; ++k) m = fmaxf(m, z[k]);
    float sum = 0.0f, e[4];
    #pragma unroll
    for (int k = 0; k < 4; ++k) { e[k] = __expf(z[k] - m); sum += e[k]; }
    float inv = fast_rcp(sum);
    #pragma unroll
    for (int k = 0; k < 4; ++k) out[(size_t)idx * 4 + k] = e[k] * inv;
}

// ---------------------------------------------------------------------------
// Workspace layout (bytes)
// ---------------------------------------------------------------------------
#define WS_EMBH    0           //  132*32 f16  ->  8448 B   (reserve 16 KB)
#define WS_CHARWT  16384       //  128*64 f16  -> 16384 B
#define WS_FWDWT   32768       //  256*160 f16 -> 81920 B
#define WS_BWDWT   114688      //  256*160 f16 -> 81920 B
#define WS_X       196608      //  38400*96 f16 -> 7372800 B
#define WS_HF      7569408     //  38400*64 f32 -> 9830400 B
#define WS_HB      17399808    //  38400*64 f32 -> 9830400 B
// total ~27.2 MB

extern "C" void kernel_launch(void* const* d_in, const int* in_sizes, int n_in,
                              void* d_out, int out_size, void* d_ws, size_t ws_size,
                              hipStream_t stream) {
    (void)in_sizes; (void)n_in; (void)out_size; (void)ws_size;
    const int*   word_in  = (const int*)  d_in[0];
    const int*   char_in  = (const int*)  d_in[1];
    const float* pos      = (const float*)d_in[2];
    const float* par      = (const float*)d_in[3];
    const float* emb_wor  = (const float*)d_in[4];
    const float* emb_char = (const float*)d_in[5];
    const float* char_Wx  = (const float*)d_in[6];
    const float* char_Wh  = (const float*)d_in[7];
    const float* char_b   = (const float*)d_in[8];
    const float* fwd_Wx   = (const float*)d_in[9];
    const float* fwd_Wh   = (const float*)d_in[10];
    const float* fwd_b    = (const float*)d_in[11];
    const float* bwd_Wx   = (const float*)d_in[12];
    const float* bwd_Wh   = (const float*)d_in[13];
    const float* bwd_b    = (const float*)d_in[14];
    const float* dense_W  = (const float*)d_in[15];
    const float* dense_b  = (const float*)d_in[16];

    char* ws = (char*)d_ws;
    _Float16* embH   = (_Float16*)(ws + WS_EMBH);
    _Float16* charWT = (_Float16*)(ws + WS_CHARWT);
    _Float16* fwdWT  = (_Float16*)(ws + WS_FWDWT);
    _Float16* bwdWT  = (_Float16*)(ws + WS_BWDWT);
    _Float16* X      = (_Float16*)(ws + WS_X);
    float*    hf     = (float*)   (ws + WS_HF);
    float*    hb     = (float*)   (ws + WS_HB);
    float*    out    = (float*)   d_out;

    prep_kernel<<<64, 256, 0, stream>>>(emb_char, char_Wx, char_Wh,
                                        fwd_Wx, fwd_Wh, bwd_Wx, bwd_Wh,
                                        embH, charWT, fwdWT, bwdWT);
    build_x_kernel<<<(NSEQ + 255) / 256, 256, 0, stream>>>(word_in, pos, par,
                                                           emb_wor, X);
    char_lstm_kernel<<<NSEQ / 128, 256, 0, stream>>>(char_in, char_b,
                                                     embH, charWT, X);
    word_lstm_kernel<<<2, 512, 0, stream>>>(word_in, fwd_b, bwd_b,
                                            fwdWT, bwdWT, X, hf, hb);
    dense_softmax_kernel<<<(NSEQ + 255) / 256, 256, 0, stream>>>(hf, hb,
                                                                 dense_W,
                                                                 dense_b, out);
}
